// grafiti__66400194396629
// MI455X (gfx1250) — matrix operand order repaired and checked by reading, not compiled, not run on hardware
//
// GraFITi forward for MI455X (gfx1250): bf16 WMMA GEMMs + fused flash attention.
// v4: branch-free A staging (clamped OOB rows, 2-step segment select), straight
//     global b128 -> LDS b128 -> v_wmma pipelines, global_prefetch for next tiles.
#include <hip/hip_runtime.h>

#define BB   16
#define TTc  128
#define DDc  41
#define NKc  128
#define NHc  4
#define NLc  3
#define KKc  (TTc*DDc)     // 5248 (divisible by 32)
#define LQCAP 128
#define DH   (NKc/NHc)     // 32
#define LDP  40            // LDS row stride in halfwords (80B, 16B-aligned rows)

typedef __attribute__((ext_vector_type(16))) __bf16 v16bf;
typedef __attribute__((ext_vector_type(8)))  float  v8f;
typedef __attribute__((ext_vector_type(8)))  unsigned short us8;

__device__ __forceinline__ float bf2f(unsigned short u){
  unsigned x = ((unsigned)u) << 16; float f; __builtin_memcpy(&f, &x, 4); return f;
}
__device__ __forceinline__ unsigned short f2bf(float f){
  unsigned x; __builtin_memcpy(&x, &f, 4);
  x = (x + 0x7FFFu + ((x >> 16) & 1u)) >> 16;
  return (unsigned short)x;
}
__device__ __forceinline__ v16bf combine16(us8 lo, us8 hi){
  v16bf f;
  __builtin_memcpy(&f, &lo, 16);
  __builtin_memcpy((char*)&f + 16, &hi, 16);
  return f;
}

// ---------------- elementwise / setup kernels ----------------
__global__ void cvt_f32_bf16(const float* __restrict__ s, unsigned short* __restrict__ d, int n){
  int i = blockIdx.x * 256 + threadIdx.x;
  if (i < n) d[i] = f2bf(s[i]);
}

// Stable partition (observed first, row-major order kept) of T*D slots per batch.
__global__ void compact_kernel(const float* __restrict__ obs, const float* __restrict__ tgt,
                               const float* __restrict__ vals,
                               float* __restrict__ maskF, float* __restrict__ tgtF,
                               float* __restrict__ valsF, int* __restrict__ tIdx,
                               int* __restrict__ cIdx, int* __restrict__ p2f){
  const int b = blockIdx.x;
  const int tid = threadIdx.x;
  __shared__ int cnt[256];
  __shared__ int total;
  const int chunk = (KKc + 255) / 256;
  int s = tid * chunk; if (s > KKc) s = KKc;
  int e = s + chunk;   if (e > KKc) e = KKc;
  const float* ob = obs  + (size_t)b * KKc;
  const float* tg = tgt  + (size_t)b * KKc;
  const float* vv = vals + (size_t)b * KKc;
  int c = 0;
  for (int i = s; i < e; ++i) if (ob[i] + tg[i] > 0.f) ++c;
  cnt[tid] = c; __syncthreads();
  for (int off = 1; off < 256; off <<= 1){
    int v = (tid >= off) ? cnt[tid - off] : 0;
    __syncthreads();
    cnt[tid] += v;
    __syncthreads();
  }
  if (tid == 255) total = cnt[255];
  __syncthreads();
  const int tot = total;
  const int obsExcl = cnt[tid] - c;
  int op = obsExcl;
  int up = tot + (s - obsExcl);
  for (int i = s; i < e; ++i){
    float m = ob[i] + tg[i];
    bool mk = m > 0.f;
    int pos = mk ? op++ : up++;
    size_t o = (size_t)b * KKc + pos;
    maskF[o] = mk ? 1.f : 0.f;
    tgtF[o]  = mk ? tg[i] : 0.f;
    valsF[o] = mk ? vv[i] : 0.f;
    tIdx[o]  = mk ? (i / DDc) : 0;
    cIdx[o]  = mk ? (i % DDc) : 0;
    p2f[o]   = i;
  }
}

__global__ void edge_init_kernel(const float* __restrict__ maskF, const float* __restrict__ tgtF,
                                 const float* __restrict__ valsF,
                                 const float* __restrict__ W, const float* __restrict__ bias,
                                 unsigned short* __restrict__ edge){
  size_t i = (size_t)blockIdx.x * 256 + threadIdx.x;
  if (i >= (size_t)BB * KKc * NKc) return;
  int n = (int)(i & (NKc - 1));
  size_t row = i >> 7;
  float mk = maskF[row];
  float ti = 1.f - mk + tgtF[row];
  float v = valsF[row] * W[n] + ti * W[NKc + n] + bias[n];
  edge[i] = f2bf(fmaxf(v, 0.f) * mk);
}

__global__ void temb_init_kernel(const float* __restrict__ tp, const float* __restrict__ W,
                                 const float* __restrict__ bias, unsigned short* __restrict__ temb){
  size_t i = (size_t)blockIdx.x * 256 + threadIdx.x;
  if (i >= (size_t)BB * TTc * NKc) return;
  int n = (int)(i & (NKc - 1));
  size_t row = i >> 7;
  temb[i] = f2bf(__sinf(tp[row] * W[n] + bias[n]));
}

__global__ void cemb_init_kernel(const float* __restrict__ W, const float* __restrict__ bias,
                                 unsigned short* __restrict__ cemb){
  size_t i = (size_t)blockIdx.x * 256 + threadIdx.x;
  if (i >= (size_t)BB * DDc * NKc) return;
  int n = (int)(i & (NKc - 1));
  int d = (int)((i >> 7) % DDc);
  cemb[i] = f2bf(fmaxf(W[d * NKc + n] + bias[n], 0.f));
}

__global__ void scatter_kernel(const float* __restrict__ outv, const int* __restrict__ p2f,
                               const float* __restrict__ tgtF, float* __restrict__ y){
  size_t i = (size_t)blockIdx.x * 256 + threadIdx.x;
  if (i >= (size_t)BB * KKc) return;
  size_t b = i / KKc;
  y[b * KKc + p2f[i]] = outv[i] * tgtF[i];
}

// ---------------- WMMA fragment loaders (vectorized: 2x ds_load_b128 each) ----------------
// A (16x32): lane L<16 holds row M=L, K=0..15; lane L+16 holds M=L, K=16..31
__device__ __forceinline__ v16bf lds_a_frag(const unsigned short* lds, int row0, int lane){
  const unsigned short* p = lds + (row0 + (lane & 15)) * LDP + ((lane >> 4) << 4);
  return combine16(*(const us8*)p, *(const us8*)(p + 8));
}
// B (32x16) from TRANSPOSED LDS tile [n][k]: lane L<16 -> col N=L, K=0..15; L+16 -> K=16..31
__device__ __forceinline__ v16bf lds_bT_frag(const unsigned short* lds, int n0, int lane){
  const unsigned short* p = lds + (n0 + (lane & 15)) * LDP + ((lane >> 4) << 4);
  return combine16(*(const us8*)p, *(const us8*)(p + 8));
}

// ---------------- general batched bf16 GEMM, gather-fused A, epilogue variants ----------------
struct Seg {
  const unsigned short* ptr;
  const int* rowmap;   // optional per-batch (stride KKc) row gather
  long bstride;        // per-batch element stride of ptr
  int ld;
  int kw;              // segment width along Kd (multiple of 32)
};
struct GemmP {
  Seg s0, s1, s2;
  const unsigned short* W;     // Kd x N row-major bf16
  const float* bias;           // length N
  void* C; long cbstride; int ldc;
  const unsigned short* R; long rbstride; int ldr;   // residual (modes 2,3)
  const float* rowscale; long rsstride;              // mode 3 per-row scale
  int M, N, Kd, mode;          // mode: 0 store, 1 relu, 2 R+relu(C), 3 relu(R+C)*scale, 5 f32 store
};

__global__ __launch_bounds__(128) void gemm_kernel(GemmP p){
  __shared__ __attribute__((aligned(16))) unsigned short As[64][LDP];
  __shared__ __attribute__((aligned(16))) unsigned short BsT[64][LDP];  // [n][k]
  const int b  = blockIdx.z;
  const int m0 = blockIdx.x * 64;
  const int n0 = blockIdx.y * 64;
  const int tid = threadIdx.x;
  const int lane = tid & 31, wave = tid >> 5;
  const bool nvec = (p.N & 7) == 0;
  v8f acc[4];
#pragma unroll
  for (int j = 0; j < 4; ++j)
#pragma unroll
    for (int v = 0; v < 8; ++v) acc[j][v] = 0.f;

  // ---- per-thread staging coordinates, resolved ONCE (k-invariant) ----
  // OOB rows (gr >= M) are CLAMPED, not zeroed: C row m depends only on A row m,
  // and rows >= M are never stored by the epilogue, so garbage there is harmless.
  const int r0  = tid >> 2;   // A row in [0,32); second row is r0+32
  const int cch = tid & 3;    // 8-halfword chunk within the 32-wide k slice
  const int kofs1 = p.s0.kw;
  const int kofs2 = p.s0.kw + p.s1.kw;
  const unsigned short *a0[2], *a1[2], *a2[2];
  {
    const Seg* sgs[3] = {&p.s0, &p.s1, &p.s2};
#pragma unroll
    for (int half = 0; half < 2; ++half){
      int gr = m0 + r0 + half * 32;
      if (gr >= p.M) gr = p.M - 1;           // clamp
      const unsigned short* bp[3];
#pragma unroll
      for (int si = 0; si < 3; ++si){
        const Seg* sg = sgs[si];
        if (sg->kw > 0){
          int row = sg->rowmap ? sg->rowmap[(size_t)b * KKc + gr] : gr;
          bp[si] = sg->ptr + (size_t)b * sg->bstride + (size_t)row * sg->ld;
        } else {
          bp[si] = nullptr;
        }
      }
      if (!bp[1]) bp[1] = bp[0];             // unused segments: any valid pointer
      if (!bp[2]) bp[2] = bp[1];
      a0[half] = bp[0]; a1[half] = bp[1]; a2[half] = bp[2];
    }
  }
  const int wkk = tid >> 3, wnch = tid & 7;      // W staging coords (nvec path)
  const int gn0w = n0 + wnch * 8;

  for (int k0 = 0; k0 < p.Kd; k0 += 32){
    const int gk = k0 + cch * 8;
    const bool ge1 = gk >= kofs1;
    const bool ge2 = gk >= kofs2;
    const int ko = gk - (ge2 ? kofs2 : (ge1 ? kofs1 : 0));
    const bool pf = (k0 + 32) < p.Kd;
    // ---- stage A (two b128 chunks per thread), branch-free ----
#pragma unroll
    for (int half = 0; half < 2; ++half){
      const unsigned short* bp = a0[half];
      if (ge1) bp = a1[half];
      if (ge2) bp = a2[half];
      *(us8*)(&As[r0 + half * 32][cch * 8]) = *(const us8*)(bp + ko);
      if (pf) __builtin_prefetch(bp + ko + 32, 0, 1);   // speculative; boundary slop is harmless
    }
    // ---- stage W transposed (BsT[n][k]) ----
    if (nvec){
#pragma unroll
      for (int half = 0; half < 2; ++half){
        const int kk = wkk + half * 16;
        us8 v8 = {0, 0, 0, 0, 0, 0, 0, 0};
        if (gn0w < p.N) v8 = *(const us8*)(p.W + (size_t)(k0 + kk) * p.N + gn0w);
#pragma unroll
        for (int e = 0; e < 8; ++e) BsT[wnch * 8 + e][kk] = v8[e];
      }
      if (pf && gn0w < p.N)
        __builtin_prefetch(p.W + (size_t)(k0 + 32 + wkk) * p.N + gn0w, 0, 1);
    } else {
      for (int l = tid; l < 32 * 64; l += 128){
        int kk = l >> 6, n = l & 63;
        int gkw = k0 + kk, gn = n0 + n;
        BsT[n][kk] = (gn < p.N) ? p.W[(size_t)gkw * p.N + gn] : (unsigned short)0;
      }
    }
    __syncthreads();
    v16bf af = lds_a_frag(&As[0][0], wave * 16, lane);
#pragma unroll
    for (int j = 0; j < 4; ++j){
      v16bf bf = lds_bT_frag(&BsT[0][0], j * 16, lane);
      acc[j] = __builtin_amdgcn_wmma_f32_16x16x32_bf16(false, af, false, bf, (short)0, acc[j], false, false);
    }
    __syncthreads();
  }
  const int mbase = m0 + wave * 16 + ((lane >> 4) << 3);
  const int nc = lane & 15;
#pragma unroll
  for (int j = 0; j < 4; ++j){
    int gn = n0 + j * 16 + nc;
    if (gn >= p.N) continue;
    float bia = p.bias ? p.bias[gn] : 0.f;
#pragma unroll
    for (int v = 0; v < 8; ++v){
      int gm = mbase + v;
      if (gm >= p.M) continue;
      float val = acc[j][v] + bia;
      size_t co = (size_t)b * p.cbstride + (size_t)gm * p.ldc + gn;
      if (p.mode == 0){
        ((unsigned short*)p.C)[co] = f2bf(val);
      } else if (p.mode == 1){
        ((unsigned short*)p.C)[co] = f2bf(fmaxf(val, 0.f));
      } else if (p.mode == 2){
        float r = bf2f(p.R[(size_t)b * p.rbstride + (size_t)gm * p.ldr + gn]);
        ((unsigned short*)p.C)[co] = f2bf(r + fmaxf(val, 0.f));
      } else if (p.mode == 3){
        float r = bf2f(p.R[(size_t)b * p.rbstride + (size_t)gm * p.ldr + gn]);
        float sc = p.rowscale[(size_t)b * p.rsstride + gm];
        ((unsigned short*)p.C)[co] = f2bf(fmaxf(r + val, 0.f) * sc);
      } else { // 5
        ((float*)p.C)[co] = val;
      }
    }
  }
}

// ---------------- fused flash attention (per 16-row q-tile per head per batch) ----------------
__global__ __launch_bounds__(128) void attn_kernel(
    const unsigned short* __restrict__ q,   // B x LQCAP x NK
    const unsigned short* __restrict__ kp,  // B x K x NK
    const unsigned short* __restrict__ vp,  // B x K x NK
    const int* __restrict__ idx,            // B x K key entity id (channel or time)
    const float* __restrict__ maskF,        // B x K
    unsigned short* __restrict__ o,         // B x LQCAP x NK : qh + softmax(S)@V
    int Lq)
{
  __shared__ __attribute__((aligned(16))) unsigned short Ks[32][LDP];     // [key][dh]
  __shared__ __attribute__((aligned(16))) unsigned short Vt[32][LDP];     // [dh][key]
  __shared__ __attribute__((aligned(16))) unsigned short Ps[4][16][LDP];  // per-wave P tile
  const int b = blockIdx.z, h = blockIdx.y;
  const int tid = threadIdx.x, lane = tid & 31, wave = tid >> 5;
  const int qrow0 = blockIdx.x * 64 + wave * 16;
  const size_t qb = (size_t)b * LQCAP * NKc;
  const size_t kb = (size_t)b * KKc * NKc;
  const float scale = 0.08838834764831845f;  // 1/sqrt(NK)

  v16bf qf;
  {
    int m = qrow0 + (lane & 15);
    if (m >= Lq) m = Lq - 1;                 // clamp; padded rows never stored
    const unsigned short* qr = q + qb + (size_t)m * NKc + h * DH + ((lane >> 4) << 4);
    qf = combine16(*(const us8*)qr, *(const us8*)(qr + 8));
  }
  float mi[8], li[8];
  v8f acc0, acc1;
#pragma unroll
  for (int v = 0; v < 8; ++v){ mi[v] = -3.0e38f; li[v] = 0.f; acc0[v] = 0.f; acc1[v] = 0.f; }
  const int rh = (lane >> 4) << 3;   // +8 rows for upper lane half (C layout)
  const int nc = lane & 15;
  // k-invariant staging coords: one b128 chunk per thread per tile
  const int skk = tid >> 2, sc8 = (tid & 3) * 8;
  const unsigned short* kpt = kp + kb + (size_t)skk * NKc + h * DH + sc8;
  const unsigned short* vpt = vp + kb + (size_t)skk * NKc + h * DH + sc8;

  for (int p0 = 0; p0 < KKc; p0 += 32){
    __syncthreads();
    // stage K rows (b128) and V transposed; prefetch next tile
    {
      const size_t s = (size_t)p0 * NKc;
      *(us8*)(&Ks[skk][sc8]) = *(const us8*)(kpt + s);
      us8 vv = *(const us8*)(vpt + s);
#pragma unroll
      for (int e = 0; e < 8; ++e) Vt[sc8 + e][skk] = vv[e];
      if (p0 + 32 < KKc){
        __builtin_prefetch(kpt + s + (size_t)32 * NKc, 0, 1);
        __builtin_prefetch(vpt + s + (size_t)32 * NKc, 0, 1);
      }
    }
    __syncthreads();

    float sv[2][8];
#pragma unroll
    for (int t = 0; t < 2; ++t){
      // B-operand = K^T: lane column = key (row of Ks), elements = contiguous dh halves
      const unsigned short* kr = &Ks[t * 16 + nc][(lane >> 4) << 4];
      v16bf kf = combine16(*(const us8*)kr, *(const us8*)(kr + 8));
      v8f z;
#pragma unroll
      for (int v = 0; v < 8; ++v) z[v] = 0.f;
      v8f s = __builtin_amdgcn_wmma_f32_16x16x32_bf16(false, qf, false, kf, (short)0, z, false, false);
      const int key = p0 + t * 16 + nc;
      const int kid = idx[(size_t)b * KKc + key];
      const bool kok = maskF[(size_t)b * KKc + key] > 0.f;
#pragma unroll
      for (int v = 0; v < 8; ++v){
        int qrow = qrow0 + rh + v;
        sv[t][v] = (kok && kid == qrow) ? s[v] * scale : -1.0e10f;
      }
    }
    // online softmax (row reductions across the 16 lanes of each half)
#pragma unroll
    for (int v = 0; v < 8; ++v){
      float mx = fmaxf(sv[0][v], sv[1][v]);
#pragma unroll
      for (int d = 1; d < 16; d <<= 1) mx = fmaxf(mx, __shfl_xor(mx, d, 16));
      float mnew = fmaxf(mi[v], mx);
      float corr = __expf(mi[v] - mnew);
      float e0 = __expf(sv[0][v] - mnew);
      float e1 = __expf(sv[1][v] - mnew);
      float rs = e0 + e1;
#pragma unroll
      for (int d = 1; d < 16; d <<= 1) rs += __shfl_xor(rs, d, 16);
      li[v] = li[v] * corr + rs;
      mi[v] = mnew;
      acc0[v] *= corr; acc1[v] *= corr;
      Ps[wave][rh + v][nc]      = f2bf(e0);
      Ps[wave][rh + v][16 + nc] = f2bf(e1);
    }
    // P @ V : A = P(16x32 keys), B = V(32 keys x 16 dh) twice (from transposed Vt)
    v16bf pf = lds_a_frag(&Ps[wave][0][0], 0, lane);
#pragma unroll
    for (int t = 0; t < 2; ++t){
      const unsigned short* vr = &Vt[t * 16 + nc][(lane >> 4) << 4];
      v16bf vf = combine16(*(const us8*)vr, *(const us8*)(vr + 8));
      if (t == 0)
        acc0 = __builtin_amdgcn_wmma_f32_16x16x32_bf16(false, pf, false, vf, (short)0, acc0, false, false);
      else
        acc1 = __builtin_amdgcn_wmma_f32_16x16x32_bf16(false, pf, false, vf, (short)0, acc1, false, false);
    }
  }
  // O = qh + acc/li
#pragma unroll
  for (int v = 0; v < 8; ++v){
    int qrow = qrow0 + rh + v;
    if (qrow >= Lq) continue;
    float inv = 1.f / li[v];
    size_t base = qb + (size_t)qrow * NKc + h * DH;
    o[base + nc]      = f2bf(bf2f(q[base + nc])      + acc0[v] * inv);
    o[base + 16 + nc] = f2bf(bf2f(q[base + 16 + nc]) + acc1[v] * inv);
  }
}

// ---------------- host orchestration ----------------
extern "C" void kernel_launch(void* const* d_in, const int* in_sizes, int n_in,
                              void* d_out, int out_size, void* d_ws, size_t ws_size,
                              hipStream_t stream){
  (void)in_sizes; (void)n_in; (void)out_size; (void)ws_size;
  auto F = [&](int i){ return (const float*)d_in[i]; };
  // inputs: 0 time, 1 values, 2 obs, 3 tgt; params (insertion order):
  // 4/5 edge_init w,b; 6/7 time_init; 8/9 chan_init; 10/11 output;
  // ct_attn[i] at 12+i*8 (q.w q.b k.w k.b v.w v.b o.w o.b); tc_attn at 36+i*8; edge_nn at 60+i*2.
  const float* time_pts = F(0);
  const float* values   = F(1);
  const float* obs      = F(2);
  const float* tgtm     = F(3);

  size_t off = 0;
  auto alloc = [&](size_t bytes)->void*{
    size_t a = (off + 255) & ~(size_t)255;
    off = a + bytes;
    return (void*)((char*)d_ws + a);
  };
  float* maskF = (float*)alloc((size_t)BB * KKc * 4);
  float* tgtF  = (float*)alloc((size_t)BB * KKc * 4);
  float* valsF = (float*)alloc((size_t)BB * KKc * 4);
  int*   tIdx  = (int*)alloc((size_t)BB * KKc * 4);
  int*   cIdx  = (int*)alloc((size_t)BB * KKc * 4);
  int*   p2f   = (int*)alloc((size_t)BB * KKc * 4);
  unsigned short* edgeA = (unsigned short*)alloc((size_t)BB * KKc * NKc * 2);
  unsigned short* edgeB = (unsigned short*)alloc((size_t)BB * KKc * NKc * 2);
  unsigned short* tembA = (unsigned short*)alloc((size_t)BB * TTc * NKc * 2);
  unsigned short* tembB = (unsigned short*)alloc((size_t)BB * TTc * NKc * 2);
  unsigned short* cembA = (unsigned short*)alloc((size_t)BB * DDc * NKc * 2);
  unsigned short* cembB = (unsigned short*)alloc((size_t)BB * DDc * NKc * 2);
  unsigned short* qbuf  = (unsigned short*)alloc((size_t)BB * LQCAP * NKc * 2);
  unsigned short* obuf  = (unsigned short*)alloc((size_t)BB * LQCAP * NKc * 2);
  unsigned short* kbuf  = (unsigned short*)alloc((size_t)BB * KKc * NKc * 2);
  unsigned short* vbuf  = (unsigned short*)alloc((size_t)BB * KKc * NKc * 2);
  float* outbuf = (float*)alloc((size_t)BB * KKc * 4);

  auto cvt = [&](const float* src, int n)->unsigned short*{
    unsigned short* d = (unsigned short*)alloc((size_t)n * 2);
    cvt_f32_bf16<<<dim3((unsigned)((n + 255) / 256)), dim3(256), 0, stream>>>(src, d, n);
    return d;
  };
  struct WBF { unsigned short *q, *k, *v, *o; };
  WBF wct[NLc], wtc[NLc];
  unsigned short* wenn[NLc];
  for (int i = 0; i < NLc; ++i){
    int bc = 12 + i * 8;
    wct[i].q = cvt(F(bc + 0), NKc * NKc);
    wct[i].k = cvt(F(bc + 2), 2 * NKc * NKc);
    wct[i].v = cvt(F(bc + 4), 2 * NKc * NKc);
    wct[i].o = cvt(F(bc + 6), NKc * NKc);
    int bt = 36 + i * 8;
    wtc[i].q = cvt(F(bt + 0), NKc * NKc);
    wtc[i].k = cvt(F(bt + 2), 2 * NKc * NKc);
    wtc[i].v = cvt(F(bt + 4), 2 * NKc * NKc);
    wtc[i].o = cvt(F(bt + 6), NKc * NKc);
    wenn[i]  = cvt(F(60 + i * 2), 3 * NKc * NKc);
  }
  unsigned short* woutw = cvt(F(10), 3 * NKc);

  compact_kernel<<<dim3(BB), dim3(256), 0, stream>>>(obs, tgtm, values, maskF, tgtF, valsF, tIdx, cIdx, p2f);
  {
    size_t n = (size_t)BB * KKc * NKc;
    edge_init_kernel<<<dim3((unsigned)((n + 255) / 256)), 256, 0, stream>>>(maskF, tgtF, valsF, F(4), F(5), edgeA);
  }
  {
    size_t n = (size_t)BB * TTc * NKc;
    temb_init_kernel<<<dim3((unsigned)((n + 255) / 256)), 256, 0, stream>>>(time_pts, F(6), F(7), tembA);
  }
  {
    size_t n = (size_t)BB * DDc * NKc;
    cemb_init_kernel<<<dim3((unsigned)((n + 255) / 256)), 256, 0, stream>>>(F(8), F(9), cembA);
  }

  auto seg = [&](const unsigned short* ptr, const int* rm, long bs, int ld, int kw){
    Seg s; s.ptr = ptr; s.rowmap = rm; s.bstride = bs; s.ld = ld; s.kw = kw; return s;
  };
  Seg z = seg(nullptr, nullptr, 0, 0, 0);
  auto gemm = [&](Seg s0, Seg s1, Seg s2, int M, int N, int Kd,
                  const unsigned short* W, const float* bias,
                  void* C, long cbs, int ldc, int mode,
                  const unsigned short* R, long rbs, int ldr,
                  const float* rsc, long rss){
    GemmP p;
    p.s0 = s0; p.s1 = s1; p.s2 = s2; p.W = W; p.bias = bias;
    p.C = C; p.cbstride = cbs; p.ldc = ldc;
    p.R = R; p.rbstride = rbs; p.ldr = ldr;
    p.rowscale = rsc; p.rsstride = rss;
    p.M = M; p.N = N; p.Kd = Kd; p.mode = mode;
    dim3 g((unsigned)((M + 63) / 64), (unsigned)((N + 63) / 64), BB);
    gemm_kernel<<<g, 128, 0, stream>>>(p);
  };

  unsigned short *eCur = edgeA, *eNxt = edgeB;
  unsigned short *tCur = tembA, *tNxt = tembB;
  unsigned short *cCur = cembA, *cNxt = cembB;
  const long EB = (long)KKc * NKc, TB = (long)TTc * NKc, CB = (long)DDc * NKc, QB = (long)LQCAP * NKc;

  for (int i = 0; i < NLc; ++i){
    int bc = 12 + i * 8, bt = 36 + i * 8;
    // ---- ct_attn: Q = c_emb, KV-src = [gather(t_emb), edge] ----
    gemm(seg(cCur, nullptr, CB, NKc, NKc), z, z, DDc, NKc, NKc,
         wct[i].q, F(bc + 1), qbuf, QB, NKc, 0, nullptr, 0, 0, nullptr, 0);
    gemm(seg(tCur, tIdx, TB, NKc, NKc), seg(eCur, nullptr, EB, NKc, NKc), z, KKc, NKc, 2 * NKc,
         wct[i].k, F(bc + 3), kbuf, EB, NKc, 0, nullptr, 0, 0, nullptr, 0);
    gemm(seg(tCur, tIdx, TB, NKc, NKc), seg(eCur, nullptr, EB, NKc, NKc), z, KKc, NKc, 2 * NKc,
         wct[i].v, F(bc + 5), vbuf, EB, NKc, 0, nullptr, 0, 0, nullptr, 0);
    attn_kernel<<<dim3((unsigned)((DDc + 63) / 64), NHc, BB), 128, 0, stream>>>(
        qbuf, kbuf, vbuf, cIdx, maskF, obuf, DDc);
    gemm(seg(obuf, nullptr, QB, NKc, NKc), z, z, DDc, NKc, NKc,
         wct[i].o, F(bc + 7), cNxt, CB, NKc, 2, obuf, QB, NKc, nullptr, 0);
    // ---- tc_attn: Q = t_emb, KV-src = [gather(OLD c_emb), edge] ----
    gemm(seg(tCur, nullptr, TB, NKc, NKc), z, z, TTc, NKc, NKc,
         wtc[i].q, F(bt + 1), qbuf, QB, NKc, 0, nullptr, 0, 0, nullptr, 0);
    gemm(seg(cCur, cIdx, CB, NKc, NKc), seg(eCur, nullptr, EB, NKc, NKc), z, KKc, NKc, 2 * NKc,
         wtc[i].k, F(bt + 3), kbuf, EB, NKc, 0, nullptr, 0, 0, nullptr, 0);
    gemm(seg(cCur, cIdx, CB, NKc, NKc), seg(eCur, nullptr, EB, NKc, NKc), z, KKc, NKc, 2 * NKc,
         wtc[i].v, F(bt + 5), vbuf, EB, NKc, 0, nullptr, 0, 0, nullptr, 0);
    attn_kernel<<<dim3((unsigned)((TTc + 63) / 64), NHc, BB), 128, 0, stream>>>(
        qbuf, kbuf, vbuf, tIdx, maskF, obuf, TTc);
    gemm(seg(obuf, nullptr, QB, NKc, NKc), z, z, TTc, NKc, NKc,
         wtc[i].o, F(bt + 7), tNxt, TB, NKc, 2, obuf, QB, NKc, nullptr, 0);
    // ---- edge update: relu(edge + [edge, t_g, c_g] @ W + b) * mask ----
    gemm(seg(eCur, nullptr, EB, NKc, NKc), seg(tCur, tIdx, TB, NKc, NKc), seg(cCur, cIdx, CB, NKc, NKc),
         KKc, NKc, 3 * NKc, wenn[i], F(61 + i * 2), eNxt, EB, NKc, 3, eCur, EB, NKc, maskF, KKc);
    unsigned short* tmp;
    tmp = eCur; eCur = eNxt; eNxt = tmp;
    tmp = tCur; tCur = tNxt; tNxt = tmp;
    tmp = cCur; cCur = cNxt; cNxt = tmp;
  }
  // output head: [edge, t_g, c_g] @ Wout + b  -> f32
  gemm(seg(eCur, nullptr, EB, NKc, NKc), seg(tCur, tIdx, TB, NKc, NKc), seg(cCur, cIdx, CB, NKc, NKc),
       KKc, 1, 3 * NKc, woutw, F(11), outbuf, (long)KKc, 1, 5, nullptr, 0, 0, nullptr, 0);
  scatter_kernel<<<dim3((unsigned)((BB * KKc + 255) / 256)), 256, 0, stream>>>(
      outbuf, p2f, tgtF, (float*)d_out);
}
